// GraphAttentionLayer_57913339019684
// MI455X (gfx1250) — compile-verified, matrix operand order checked by reading, not run on hardware
//
#include <hip/hip_runtime.h>
#include <hip/hip_bf16.h>

#define NN 6144
#define INF_ 256
#define OUTF 64

typedef __attribute__((ext_vector_type(16))) __bf16 v16bf;
typedef __attribute__((ext_vector_type(8)))  __bf16 v8bf;
typedef __attribute__((ext_vector_type(8)))  float  v8f;
typedef __attribute__((ext_vector_type(2)))  float  v2f;
typedef __attribute__((ext_vector_type(4)))  unsigned int v4u;
typedef __attribute__((ext_vector_type(8)))  int    v8i;
typedef __attribute__((ext_vector_type(4)))  int    v4i;

#if __has_builtin(__builtin_amdgcn_tensor_load_to_lds)
#define HAVE_TDM 1
#else
#define HAVE_TDM 0
#endif

// LDS layout (dwords): [0,1024) redAcc | [1024,1040) redSum |
//                      [1040 + wv*5280 + sel*2640 ...) edge buffers (padded rows)
#define LDS_RED_DW   1040
#define BUF_DW       2640           // 16 rows * 165 dw (160 data + 5 pad)
#define WAVE_BUF_DW  (2 * BUF_DW)
#define SMEM_DW      (LDS_RED_DW + 8 * WAVE_BUF_DW)   // 43280 dw = 173,120 B

// ---------------- Kernel 1: wh = lstm_out @ W  (WMMA f32 16x16x4) -----------
#if __has_builtin(__builtin_amdgcn_wmma_f32_16x16x4_f32)
__global__ __launch_bounds__(128) void k_wh_gemm(const float* __restrict__ X,
                                                 const float* __restrict__ W,
                                                 float* __restrict__ wh) {
  const int i0   = blockIdx.x * 16;
  const int wv   = threadIdx.x >> 5;
  const int lane = threadIdx.x & 31;
  const int r    = lane & 15;
  const int half = lane >> 4;
  const int n    = wv * 16 + r;

  v8f c = {};
  for (int k = 0; k < INF_; k += 4) {
    const int kk = k + half * 2;
    v2f a, b;
    a.x = X[(i0 + r) * INF_ + kk];
    a.y = X[(i0 + r) * INF_ + kk + 1];
    b.x = W[kk * OUTF + n];
    b.y = W[(kk + 1) * OUTF + n];
    c = __builtin_amdgcn_wmma_f32_16x16x4_f32(false, a, false, b,
                                              (short)0, c, false, false);
  }
#pragma unroll
  for (int v = 0; v < 8; ++v)
    wh[(i0 + half * 8 + v) * OUTF + n] = c[v];
}
#else
__global__ __launch_bounds__(128) void k_wh_gemm(const float* __restrict__ X,
                                                 const float* __restrict__ W,
                                                 float* __restrict__ wh) {
  const int i0 = blockIdx.x * 16;
  for (int o = threadIdx.x; o < 16 * OUTF; o += blockDim.x) {
    const int row = i0 + (o >> 6), col = o & 63;
    float s = 0.f;
    for (int k = 0; k < INF_; ++k) s += X[row * INF_ + k] * W[k * OUTF + col];
    wh[row * OUTF + col] = s;
  }
}
#endif

// ------------- Kernel 2: e_i, e_j, whT (bf16, transposed 64 x 6144) ---------
__global__ __launch_bounds__(256) void k_prep(const float* __restrict__ wh,
                                              const float* __restrict__ a,
                                              float* __restrict__ e_i,
                                              float* __restrict__ e_j,
                                              __bf16* __restrict__ whT) {
  const int i = blockIdx.x * blockDim.x + threadIdx.x;
  if (i >= NN) return;
  const float* row = wh + (size_t)i * OUTF;
  float s1 = 0.f, s3 = 0.f;
#pragma unroll 8
  for (int c = 0; c < OUTF; ++c) {
    const float v = row[c];
    s1 += v * a[c];
    s3 += v * a[69 + c];
    whT[(size_t)c * NN + i] = (__bf16)v;
  }
  e_i[i] = s1;
  e_j[i] = s3;
}

#if HAVE_TDM
// Issue one TDM 2D tile load: 16 rows x 160 f32 (one 32-j edge chunk),
// row stride = NN*5 elements, padded +1 dword per 32 dwords in LDS.
__device__ __forceinline__ void tdm_edges_chunk(unsigned long long ga,
                                                unsigned ldsAddr) {
  v4u g0;
  g0[0] = 1u;                                   // count=1, user mode
  g0[1] = ldsAddr;                              // LDS byte address
  g0[2] = (unsigned)ga;                         // global_addr lo
  g0[3] = ((unsigned)(ga >> 32) & 0x1FFFFFFu) | (2u << 30);  // hi | type=2
  v8i g1;
  g1[0] = (2 << 16) | (1 << 20) | (4 << 22);    // data_size=4B, pad_en, every 32dw +1dw
  g1[1] = (int)((NN * 5) << 16);                // tensor_dim0 = 30720 (lo16)
  g1[2] = (int)(NN << 16);                      // tensor_dim1 = 6144 (lo16); dim0 hi16=0
  g1[3] = (int)(160 << 16);                     // tile_dim0 = 160; dim1 hi16=0
  g1[4] = 16;                                   // tile_dim1 = 16, tile_dim2 = 0
  g1[5] = NN * 5;                               // tensor_dim0_stride lo32
  g1[6] = 0;
  g1[7] = 0;                                    // tensor_dim1_stride = 0
  v4i gz4 = {};
  v8i gz8 = {};
  // 6-arg toolchain form: (g0, g1, g2, g3, <extra int32x8>, cpol)
  __builtin_amdgcn_tensor_load_to_lds(g0, g1, gz4, gz4, gz8, 0);
}
#endif

// ------------- Kernel 3: fused scores + softmax + attention @ wh ------------
__global__ __launch_bounds__(256) void k_attn(const float* __restrict__ edges,
                                              const int*   __restrict__ adj,
                                              const float* __restrict__ e_i,
                                              const float* __restrict__ e_j,
                                              const float* __restrict__ a2,
                                              const __bf16* __restrict__ whT,
                                              const float* __restrict__ wh,
                                              float* __restrict__ out) {
  extern __shared__ float smem[];

  const int i0   = blockIdx.x * 16;
  const int tid  = threadIdx.x;
  const int wvu  = __builtin_amdgcn_readfirstlane(tid >> 5);  // wave id (uniform)
  const int lane = tid & 31;
  const int r    = lane & 15;
  const int half = lane >> 4;

  for (int o = tid; o < 16 * OUTF; o += 256) smem[o] = 0.f;
  if (tid < 16) smem[1024 + tid] = 0.f;
  __syncthreads();

  const float c0 = a2[0], c1 = a2[1], c2 = a2[2], c3 = a2[3], c4 = a2[4];
  const int   row  = i0 + r;
  const float ei   = e_i[row];
  const int*  arow = adj + (size_t)row * NN;
#if !HAVE_TDM
  const float* erow = edges + (size_t)row * NN * 5;
#endif

  v8f acc[4] = {{}, {}, {}, {}};
  float sumP = 0.f;

#if HAVE_TDM
  const unsigned ldsBase =
      (unsigned)(size_t)smem + (unsigned)(LDS_RED_DW + wvu * WAVE_BUF_DW) * 4u;
  const unsigned long long gaBase =
      (unsigned long long)(size_t)edges + (unsigned long long)i0 * NN * 20ull;
  // prime buffer 0 with this wave's first chunk
  tdm_edges_chunk(gaBase + (unsigned long long)(wvu * 32) * 20ull, ldsBase);
#endif

  int it = 0;
  for (int c = wvu; c < NN / 32; c += 8, ++it) {
    const int j0 = c * 32;

#if HAVE_TDM
    const int cn = c + 8;
    if (cn < NN / 32) {
      tdm_edges_chunk(gaBase + (unsigned long long)(cn * 32) * 20ull,
                      ldsBase + (unsigned)(((it + 1) & 1) * BUF_DW) * 4u);
      __builtin_amdgcn_s_wait_tensorcnt((short)1);
    } else {
      __builtin_amdgcn_s_wait_tensorcnt((short)0);
    }
    asm volatile("" ::: "memory");
    const float* ebr = smem + (LDS_RED_DW + wvu * WAVE_BUF_DW +
                               (it & 1) * BUF_DW) + 165 * r;  // this lane's row
    // prefetch next chunk's adjacency
    if (cn < NN / 32) __builtin_prefetch(arow + cn * 32 + half * 8, 0, 3);
#else
    if (c + 8 < NN / 32)
      __builtin_prefetch(erow + (size_t)((c + 8) * 32 + half * 8) * 5, 0, 3);
#endif

    // ---- build P tile (16x32 bf16) directly in the A-operand layout ----
    v16bf p = {};
    float ps = 0.f;
#pragma unroll
    for (int u = 0; u < 16; ++u) {
      const int q  = (u < 8) ? (half * 8 + u) : (16 + half * 8 + (u - 8));
      const int jj = j0 + q;
#if HAVE_TDM
      const int d0 = q * 5;
      const float x0 = ebr[d0 + 0 + ((d0 + 0) >> 5)];
      const float x1 = ebr[d0 + 1 + ((d0 + 1) >> 5)];
      const float x2 = ebr[d0 + 2 + ((d0 + 2) >> 5)];
      const float x3 = ebr[d0 + 3 + ((d0 + 3) >> 5)];
      const float x4 = ebr[d0 + 4 + ((d0 + 4) >> 5)];
#else
      const float* e5 = erow + (size_t)jj * 5;
      const float x0 = e5[0], x1 = e5[1], x2 = e5[2], x3 = e5[3], x4 = e5[4];
#endif
      const float rr = x0 * c0 + x1 * c1 + x2 * c2 + x3 * c3 + x4 * c4;
      float s = ei + rr + e_j[jj];
      s = (s > 0.f) ? s : 0.2f * s;                       // leaky_relu
      const float pv = (arow[jj] > 0) ? __expf(s) : 0.f;  // mask -> p = 0
      ps += pv;
      p[u] = (__bf16)pv;
    }
    sumP += ps;

    // ---- 4 WMMAs: D(16x16) += P(16x32) @ whT-tile(32x16), bf16 ----
    const int jb1 = j0 + half * 8;
    const int jb2 = j0 + 16 + half * 8;
#pragma unroll
    for (int nt = 0; nt < 4; ++nt) {
      const __bf16* base = whT + (size_t)(nt * 16 + r) * NN;
      const v8bf blo = *(const v8bf*)(base + jb1);
      const v8bf bhi = *(const v8bf*)(base + jb2);
      const v16bf b = __builtin_shufflevector(blo, bhi, 0, 1, 2, 3, 4, 5, 6, 7,
                                              8, 9, 10, 11, 12, 13, 14, 15);
      acc[nt] = __builtin_amdgcn_wmma_f32_16x16x32_bf16(false, p, false, b,
                                                        (short)0, acc[nt],
                                                        false, false);
    }
  }

  // ---- combine the 8 waves' partials through LDS (ds_add_f32) ----
  atomicAdd(&smem[1024 + r], sumP);
#pragma unroll
  for (int nt = 0; nt < 4; ++nt)
#pragma unroll
    for (int v = 0; v < 8; ++v)
      atomicAdd(&smem[(half * 8 + v) * OUTF + nt * 16 + r], acc[nt][v]);
  __syncthreads();

  // ---- normalize + residual ----
  for (int o = tid; o < 16 * OUTF; o += 256) {
    const int rw = o >> 6, col = o & 63;
    const int gi = i0 + rw;
    const float den = smem[1024 + rw];
    const float val = (den > 0.f) ? (smem[o] / den) : 0.f;
    out[(size_t)gi * OUTF + col] = val + wh[(size_t)gi * OUTF + col];
  }
}

// ---------------------------------------------------------------------------
extern "C" void kernel_launch(void* const* d_in, const int* in_sizes, int n_in,
                              void* d_out, int out_size, void* d_ws, size_t ws_size,
                              hipStream_t stream) {
  // inputs: 0 ids, 1 lstm_out, 2 edges_list, 3 adj_mat, 4 first, 5 W, 6 a
  const float* lstm  = (const float*)d_in[1];
  const float* edges = (const float*)d_in[2];
  const int*   adjm  = (const int*)d_in[3];
  const float* W     = (const float*)d_in[5];
  const float* a     = (const float*)d_in[6];
  float* out = (float*)d_out;

  float*  wh  = (float*)d_ws;
  float*  e_i = wh + (size_t)NN * OUTF;
  float*  e_j = e_i + NN;
  __bf16* whT = (__bf16*)(e_j + NN);

  k_wh_gemm<<<NN / 16, 128, 0, stream>>>(lstm, W, wh);
  k_prep<<<NN / 256, 256, 0, stream>>>(wh, a, e_i, e_j, whT);
  k_attn<<<NN / 16, 256, (size_t)SMEM_DW * 4, stream>>>(edges, adjm, e_i, e_j,
                                                        a + 64, whT, wh, out);
}
